// Model_15908558865616
// MI455X (gfx1250) — matrix operand order repaired and checked by reading, not compiled, not run on hardware
//
#include <hip/hip_runtime.h>
#include <hip/hip_bf16.h>

typedef __attribute__((ext_vector_type(16))) _Float16 v16h;
typedef __attribute__((ext_vector_type(8)))  _Float16 v8h;
typedef __attribute__((ext_vector_type(8)))  float    v8f;
typedef __attribute__((ext_vector_type(4)))  float    v4f;

namespace {
constexpr int kB = 16, kS = 2048, kD = 128;
constexpr int kWaves = 8;      // waves per block
constexpr int kQT = 16;        // queries per wave
constexpr int kKT = 64;        // keys per iteration
constexpr int kKStride = 136;  // halves; 272B rows: 16B-aligned, bank-rotated
constexpr int kVStride = 72;   // halves; 144B rows: 16B-aligned, bank-rotated
constexpr float kLog2e = 1.4426950408889634f;
}

__global__ __launch_bounds__(kWaves * 32, 1)
void fa_fwd_kernel(const float* __restrict__ Qg, const float* __restrict__ Kg,
                   const float* __restrict__ Vg, const float* __restrict__ Sg,
                   float* __restrict__ Og)
{
  __shared__ _Float16 Klds[kKT][kKStride];   // ~17.4 KB: K tile, row-major f16
  __shared__ _Float16 Vt[kD][kVStride];      // ~18.4 KB: V tile, transposed f16

  const int tid  = threadIdx.x;
  const int lane = tid & 31;
  const int wave = tid >> 5;
  const int qn   = lane & 15;   // query column owned by this lane (C/B layout)
  const int hi   = lane >> 4;   // 0: lanes 0-15, 1: lanes 16-31

  const int b     = blockIdx.x >> 4;
  const int qbase = (blockIdx.x & 15) * (kWaves * kQT) + wave * kQT;

  const float* Qb = Qg + ((size_t)b * kS + qbase) * kD;
  const float* Kb = Kg + (size_t)b * kS * kD;
  const float* Vb = Vg + (size_t)b * kS * kD;

  // ---- Q tile in B-operand layout: n = query (lane%16), K-dim over d ----
  v16h qmat[4];
  {
    const float* qrow = Qb + qn * kD;
    #pragma unroll
    for (int c = 0; c < 4; ++c) {
      const int d0 = c * 32 + hi * 16;
      #pragma unroll
      for (int j = 0; j < 16; ++j)
        qmat[c][j] = (_Float16)qrow[d0 + j];
    }
  }

  v8f o[8] = {};                 // O accumulators: 8 d-tiles of 16x16 f32
  float mrow = -1e30f, lrow = 0.f;   // running log2-domain max / sum

  // staging: each thread stages a PAIR of key rows x 16 d-columns (of K and V)
  const int r0 = (tid >> 3) * 2;        // 0,2,...,62
  const int cb = (tid & 7) * 16;        // 0..112

  for (int kb = 0; kb < kS; kb += kKT) {
    // ---- stage K (row-major f16) and V (transposed, f16 pairs packed b32) ----
    {
      const float* k0 = Kb + (size_t)(kb + r0) * kD + cb;
      const float* k1 = k0 + kD;
      const float* v0 = Vb + (size_t)(kb + r0) * kD + cb;
      const float* v1 = v0 + kD;
      #pragma unroll
      for (int j = 0; j < 16; ++j) {
        Klds[r0][cb + j]     = (_Float16)k0[j];
        Klds[r0 + 1][cb + j] = (_Float16)k1[j];
      }
      #pragma unroll
      for (int j = 0; j < 16; ++j) {
        union { unsigned u; _Float16 h[2]; } p;
        p.h[0] = (_Float16)v0[j];
        p.h[1] = (_Float16)v1[j];
        *(unsigned*)&Vt[cb + j][r0] = p.u;
      }
      if (kb + kKT < kS) {             // prefetch next tile (global_prefetch_b8)
        __builtin_prefetch(k0 + (size_t)kKT * kD, 0, 1);
        __builtin_prefetch(k1 + (size_t)kKT * kD, 0, 1);
        __builtin_prefetch(v0 + (size_t)kKT * kD, 0, 1);
        __builtin_prefetch(v1 + (size_t)kKT * kD, 0, 1);
      }
    }
    __syncthreads();

    // ---- S^T = K_tile · Q^T : c-outer / t-inner so 4 WMMAs share B=qmat[c] ----
    v8f st[4];
    #pragma unroll
    for (int c = 0; c < 4; ++c) {
      #pragma unroll
      for (int t = 0; t < 4; ++t) {
        const _Float16* ap = &Klds[t * 16 + qn][c * 32 + hi * 16];
        const v8h alo = *(const v8h*)(ap);
        const v8h ahi = *(const v8h*)(ap + 8);
        v16h a;
        #pragma unroll
        for (int j = 0; j < 8; ++j) { a[j] = alo[j]; a[8 + j] = ahi[j]; }
        v8f cin = (c == 0) ? (v8f){} : st[t];
        st[t] = __builtin_amdgcn_wmma_f32_16x16x32_f16(false, a, false, qmat[c],
                                                       (short)0, cin, false, false);
      }
    }

    // ---- elementwise scale, folded with log2(e) for base-2 softmax ----
    const v4f* sp = (const v4f*)(Sg + (size_t)(qbase + qn) * kS + kb + 8 * hi);
    #pragma unroll
    for (int t = 0; t < 4; ++t) {
      const v4f s0 = sp[t * 4]     * kLog2e;
      const v4f s1 = sp[t * 4 + 1] * kLog2e;
      #pragma unroll
      for (int r = 0; r < 8; ++r)
        st[t][r] *= (r < 4) ? s0[r] : s1[r - 4];
    }

    // ---- online softmax (log2 domain): lane owns query qn ----
    float lmax = st[0][0];
    #pragma unroll
    for (int t = 0; t < 4; ++t)
      #pragma unroll
      for (int r = 0; r < 8; ++r) lmax = fmaxf(lmax, st[t][r]);
    lmax = fmaxf(lmax, __shfl_xor(lmax, 16, 32));
    const bool  bump  = lmax > mrow;
    const float mnew  = fmaxf(mrow, lmax);
    const float alpha = exp2f(mrow - mnew);   // exactly 1.0 when !bump
    float psum = 0.f;
    v16h pa0, pa1;   // P in f16 A-operand layout for keys 0-31 / 32-63
    #pragma unroll
    for (int t = 0; t < 2; ++t)
      #pragma unroll
      for (int r = 0; r < 8; ++r) {
        const float p = exp2f(st[t][r] - mnew);
        psum += p;
        pa0[t * 8 + r] = (_Float16)p;
      }
    #pragma unroll
    for (int t = 2; t < 4; ++t)
      #pragma unroll
      for (int r = 0; r < 8; ++r) {
        const float p = exp2f(st[t][r] - mnew);
        psum += p;
        pa1[(t - 2) * 8 + r] = (_Float16)p;
      }
    psum += __shfl_xor(psum, 16, 32);
    lrow = lrow * alpha + psum;

    // ---- rescale O only if some row max actually moved (wave-uniform skip) ----
    if (__ballot(bump) != 0ull) {
      float ar[8];
      #pragma unroll
      for (int r = 0; r < 8; ++r) ar[r] = __shfl(alpha, 8 * hi + r, 32);
      #pragma unroll
      for (int t = 0; t < 8; ++t)
        #pragma unroll
        for (int r = 0; r < 8; ++r) o[t][r] *= ar[r];
    }
    mrow = mnew;

    // ---- O += P · V : kk-outer so 8 consecutive WMMAs share A = pa0 / pa1 ----
    #pragma unroll
    for (int kk = 0; kk < 2; ++kk) {
      const v16h& pa = kk ? pa1 : pa0;
      #pragma unroll
      for (int t = 0; t < 8; ++t) {
        const _Float16* vp = &Vt[t * 16 + qn][kk * 32 + hi * 8];
        const v8h blo = *(const v8h*)(vp);
        const v8h bh2 = *(const v8h*)(vp + 16);
        v16h bv;
        #pragma unroll
        for (int j = 0; j < 8; ++j) { bv[j] = blo[j]; bv[8 + j] = bh2[j]; }
        o[t] = __builtin_amdgcn_wmma_f32_16x16x32_f16(false, pa, false, bv,
                                                      (short)0, o[t], false, false);
      }
    }
    __syncthreads();
  }

  // ---- normalize by row sum and store ----
  const float linv = 1.f / lrow;
  float lr[8];
  #pragma unroll
  for (int r = 0; r < 8; ++r) lr[r] = __shfl(linv, 8 * hi + r, 32);

  float* orow = Og + ((size_t)b * kS + qbase) * kD;
  #pragma unroll
  for (int t = 0; t < 8; ++t)
    #pragma unroll
    for (int r = 0; r < 8; ++r)
      orow[(size_t)(r + 8 * hi) * kD + (t * 16 + qn)] = o[t][r] * lr[r];
}

extern "C" void kernel_launch(void* const* d_in, const int* in_sizes, int n_in,
                              void* d_out, int out_size, void* d_ws, size_t ws_size,
                              hipStream_t stream) {
  (void)in_sizes; (void)n_in; (void)out_size; (void)d_ws; (void)ws_size;
  const float* Q  = (const float*)d_in[0];
  const float* K  = (const float*)d_in[1];
  const float* V  = (const float*)d_in[2];
  const float* Sc = (const float*)d_in[3];
  float* O = (float*)d_out;
  dim3 grid(kB * (kS / (kWaves * kQT)));   // 16 batches * 16 q-blocks = 256
  dim3 block(kWaves * 32);                 // 8 waves (wave32)
  fa_fwd_kernel<<<grid, block, 0, stream>>>(Q, K, V, Sc, O);
}
